// SkeletonGCNEncoderLSTM_2843268350833
// MI455X (gfx1250) — matrix operand order repaired and checked by reading, not compiled, not run on hardware
//
#include <hip/hip_runtime.h>
#include <hip/hip_bf16.h>
#include <stdint.h>

typedef __attribute__((ext_vector_type(16))) _Float16 v16h;
typedef __attribute__((ext_vector_type(8)))  _Float16 v8h;
typedef __attribute__((ext_vector_type(8)))  float    v8f;
typedef __attribute__((ext_vector_type(4)))  unsigned int u32x4;
typedef __attribute__((ext_vector_type(8)))  int      i32x8;
typedef __attribute__((ext_vector_type(4)))  int      i32x4;

#define B_  64
#define T_  512
#define J_  33
#define D_  3
#define H1_ 32
#define H2_ 64
#define HL_ 128
#define NG_ 512           // 4*HL gates
#define MROWS (B_*T_)     // 32768

// ---------------- workspace layout (bytes) ----------------
#define WS_WIH_H  0            // 512*64  f16 = 65536
#define WS_WHH_H  65536        // 512*128 f16 = 131072
#define WS_BIAS   196608       // 512 f32 = 2048
#define WS_FEATS  262144       // 32768*64 f16 = 4 MB
#define WS_GIN    8388608      // [T][B][512] f16 = 32 MB

#if defined(__has_builtin)
#if __has_builtin(__builtin_amdgcn_tensor_load_to_lds) && __has_builtin(__builtin_amdgcn_s_wait_tensorcnt)
#define HAVE_TDM 1
#endif
#endif

__device__ __forceinline__ float sigmoidf_(float x) {
  return 1.0f / (1.0f + __expf(-x));
}
__device__ __forceinline__ float tanhf_(float x) {
  float e = __expf(-2.0f * x);
  return (1.0f - e) / (1.0f + e);
}

// =====================================================================
// Kernel 0: weight prep — f32 -> f16 weights, fused bias
// =====================================================================
__global__ void prep_kernel(const float* __restrict__ wih, const float* __restrict__ whh,
                            const float* __restrict__ bih, const float* __restrict__ bhh,
                            _Float16* __restrict__ wih_h, _Float16* __restrict__ whh_h,
                            float* __restrict__ bias) {
  int i = blockIdx.x * blockDim.x + threadIdx.x;      // grid covers 65536
  if (i < NG_ * H2_) wih_h[i] = (_Float16)wih[i];
  if (i < NG_ * HL_) whh_h[i] = (_Float16)whh[i];
  if (i < NG_)       bias[i]  = bih[i] + bhh[i];
}

// =====================================================================
// Kernel 1: fused GCN + joint-mean -> feats f16 [B*T, 64]
//   feats = ((1/J) colsum(adj) * relu((adj@x)@W1+b1)) @ W2 + b2
// =====================================================================
__global__ __launch_bounds__(256) void gcn_kernel(
    const float* __restrict__ x, const float* __restrict__ adj,
    const float* __restrict__ W1, const float* __restrict__ b1,
    const float* __restrict__ W2, const float* __restrict__ b2,
    _Float16* __restrict__ feats) {
  __shared__ float adjs[J_ * J_];
  __shared__ float aw[J_];
  __shared__ float w1s[D_ * H1_];
  __shared__ float b1s[H1_];
  __shared__ float w2s[H1_ * H2_];
  __shared__ float b2s[H2_];
  __shared__ float xs[8 * J_ * D_];
  __shared__ float ys[8 * J_ * D_];
  __shared__ float ms[8 * H1_];

  const int tid = threadIdx.x;
  const int p0 = blockIdx.x * 8;   // 8 (b,t) pairs per block

  for (int i = tid; i < J_ * J_; i += 256) adjs[i] = adj[i];
  for (int i = tid; i < D_ * H1_; i += 256) w1s[i] = W1[i];
  if (tid < H1_) b1s[tid] = b1[tid];
  for (int i = tid; i < H1_ * H2_; i += 256) w2s[i] = W2[i];
  if (tid < H2_) b2s[tid] = b2[tid];
  const float* xg = x + (size_t)p0 * (J_ * D_);
  for (int i = tid; i < 8 * J_ * D_; i += 256) xs[i] = xg[i];
  __syncthreads();

  // column weights (1/J)*colsum(adj), and y = adj @ x per pair
  if (tid < J_) {
    float s = 0.f;
    for (int k = 0; k < J_; ++k) s += adjs[k * J_ + tid];
    aw[tid] = s * (1.0f / (float)J_);
  }
  for (int idx = tid; idx < 8 * J_ * D_; idx += 256) {
    int pi = idx / (J_ * D_), r = idx % (J_ * D_), j = r / D_, d = r % D_;
    float s = 0.f;
    for (int k = 0; k < J_; ++k) s += adjs[j * J_ + k] * xs[pi * (J_ * D_) + k * D_ + d];
    ys[idx] = s;
  }
  __syncthreads();

  // m_c = sum_j aw[j] * relu(y_j @ W1 + b1)   (32 threads per pair)
  {
    int pi = tid >> 5, c = tid & 31;
    float m = 0.f;
    for (int j = 0; j < J_; ++j) {
      float t = b1s[c];
      const float* yj = &ys[pi * (J_ * D_) + j * D_];
      for (int d = 0; d < D_; ++d) t += yj[d] * w1s[d * H1_ + c];
      t = fmaxf(t, 0.f);
      m += aw[j] * t;
    }
    ms[pi * H1_ + c] = m;
  }
  __syncthreads();

  // feats = m @ W2 + b2
  for (int idx = tid; idx < 8 * H2_; idx += 256) {
    int pi = idx >> 6, n = idx & 63;
    float f = b2s[n];
    for (int c = 0; c < H1_; ++c) f += ms[pi * H1_ + c] * w2s[c * H2_ + n];
    feats[(size_t)(p0 + pi) * H2_ + n] = (_Float16)f;
  }
}

// =====================================================================
// Kernel 2: Gin GEMM (WMMA):  Gin[t*64+b, n] = feats[b*512+t, :] @ w_ih^T
//   M=32768, N=512, K=64 ; f16 in, f16 out (pre-activation, no bias)
// =====================================================================
__global__ __launch_bounds__(256) void gin_gemm_kernel(
    const _Float16* __restrict__ A,   // feats [M,64] row-major
    const _Float16* __restrict__ Bm,  // w_ih f16 [512,64] = [N,K] row-major
    _Float16* __restrict__ Gin) {     // [T][B][512]
  const int tid = threadIdx.x, lane = tid & 31, w = tid >> 5;
  const int bm = blockIdx.x >> 2, nq = blockIdx.x & 3;
  const int mBase = bm * 128 + w * 16;
  const int nBase = nq * 128;
  const int laneLo = lane & 15;
  const int hiA = (lane & 16) ? 8 : 0;
  const int hiB = (lane & 16) ? 16 : 0;

  v8f acc[8];
#pragma unroll
  for (int j = 0; j < 8; ++j)
#pragma unroll
    for (int r = 0; r < 8; ++r) acc[j][r] = 0.f;

#pragma unroll
  for (int kt = 0; kt < 2; ++kt) {
    // A-tile (16x32 f16): halfs 0-15 = K base+16h.. (contiguous)
    const _Float16* pa = A + (size_t)(mBase + laneLo) * H2_ + kt * 32 + hiB;
    v16h a;
#pragma unroll
    for (int i = 0; i < 8; ++i) a[i] = pa[i];
#pragma unroll
    for (int i = 0; i < 8; ++i) a[8 + i] = pa[8 + i];
#pragma unroll
    for (int j = 0; j < 8; ++j) {
      const _Float16* pb = Bm + (size_t)(nBase + j * 16 + laneLo) * H2_ + kt * 32 + hiB;
      v16h b;
#pragma unroll
      for (int i = 0; i < 16; ++i) b[i] = pb[i];
      acc[j] = __builtin_amdgcn_wmma_f32_16x16x32_f16(false, a, false, b, (short)0,
                                                      acc[j], false, false);
    }
  }
  // epilogue: scatter into time-major layout
#pragma unroll
  for (int j = 0; j < 8; ++j) {
    int n = nBase + j * 16 + laneLo;
#pragma unroll
    for (int r = 0; r < 8; ++r) {
      int m = mBase + hiA + r;          // C layout: VGPR r -> M=r (lo lanes) / r+8 (hi)
      int t = m & (T_ - 1);
      int b = m >> 9;
      Gin[(size_t)(t * B_ + b) * NG_ + n] = (_Float16)acc[j][r];
    }
  }
}

// =====================================================================
// Kernel 3: persistent fused LSTM (1 workgroup, 16 waves, WMMA + TDM)
// =====================================================================
#define GIN_OFF    0u
#define GIN_BYTES  (B_ * NG_ * 2)                 // 65536
#define GATES_OFF  (2u * GIN_BYTES)               // 131072
#define GSTRIDE    (NG_ + 8)                      // padded f16 row
#define H_OFF      (GATES_OFF + B_ * GSTRIDE * 2) // 197632
#define HSTRIDE    (HL_ + 8)                      // padded f16 row
#define SMEM_TOTAL (H_OFF + B_ * HSTRIDE * 2)     // 215040 bytes

__device__ __forceinline__ void tdm_issue_gin(unsigned int lds_off, const _Float16* src) {
#if defined(HAVE_TDM)
  unsigned long long ga = (unsigned long long)(uintptr_t)src;
  u32x4 g0;
  g0[0] = 1u;                                          // count=1, user descriptor
  g0[1] = lds_off;                                     // lds_addr
  g0[2] = (unsigned int)ga;                            // global_addr[31:0]
  g0[3] = ((unsigned int)(ga >> 32) & 0x01FFFFFFu) | 0x80000000u; // addr hi + type=2
  i32x8 g1;
  g1[0] = (int)0x00010000;  // workgroup_mask=0, data_size=1 (2 bytes)
  g1[1] = (int)0x80000000;  // tensor_dim0 lo16 = 0x8000 (32768 elems) in [31:16]
  g1[2] = (int)0x00010000;  // tensor_dim0 hi=0 ; tensor_dim1 lo16 = 1
  g1[3] = (int)0x80000000;  // tensor_dim1 hi=0 ; tile_dim0 = 0x8000
  g1[4] = 1;                // tile_dim1 = 1, tile_dim2 = 0
  g1[5] = 32768;            // tensor_dim0_stride lo32
  g1[6] = (int)0x80000000;  // stride hi=0 ; tensor_dim1_stride lo16 = 0x8000
  g1[7] = 0;
  i32x4 z4 = {0, 0, 0, 0};
  i32x8 z8 = {0, 0, 0, 0, 0, 0, 0, 0};
  __builtin_amdgcn_tensor_load_to_lds(g0, g1, z4, z4, z8, 0);
#else
  (void)lds_off; (void)src;
#endif
}

__global__ __launch_bounds__(512) void lstm_kernel(
    const _Float16* __restrict__ Gin,   // [T][B][512] f16
    const _Float16* __restrict__ Whh,   // f16 [512,128] = [N,K] row-major
    const float* __restrict__ bias,     // [512]
    float* __restrict__ out) {          // [B,128]
  extern __shared__ char smem[];
  _Float16* ginb = (_Float16*)(smem + GIN_OFF);
  _Float16* gts  = (_Float16*)(smem + GATES_OFF);
  _Float16* hsm  = (_Float16*)(smem + H_OFF);

  const int tid = threadIdx.x, lane = tid & 31, wid = tid >> 5;
  const int laneLo = lane & 15;
  const int hiA = (lane & 16) ? 8 : 0;
  const int hiB = (lane & 16) ? 16 : 0;

  // persistent w_hh B-tiles in VGPRs: wave owns N-tiles {2w, 2w+1}, 4 K-tiles
  v16h Breg[2][4];
  float biasv[2];
#pragma unroll
  for (int j = 0; j < 2; ++j) {
    int n = (wid * 2 + j) * 16 + laneLo;
    biasv[j] = bias[n];
#pragma unroll
    for (int kt = 0; kt < 4; ++kt) {
      const _Float16* pb = Whh + (size_t)n * HL_ + kt * 32 + hiB;
      v16h b;
#pragma unroll
      for (int i = 0; i < 16; ++i) b[i] = pb[i];
      Breg[j][kt] = b;
    }
  }

  for (int i = tid; i < B_ * HSTRIDE; i += 512) hsm[i] = (_Float16)0.f;
  float cst[16];
#pragma unroll
  for (int i = 0; i < 16; ++i) cst[i] = 0.f;
  const int mrow = tid >> 3;
  const int hb = (tid & 7) * 16;

#if defined(HAVE_TDM)
  if (wid == 0) {
    tdm_issue_gin(GIN_OFF, Gin);               // preload t=0 into buffer 0
    __builtin_amdgcn_s_wait_tensorcnt(0);
  }
#endif
  __syncthreads();

  for (int t = 0; t < T_; ++t) {
#if defined(HAVE_TDM)
    const int cur = t & 1;
    if (wid == 0 && (t + 1) < T_)
      tdm_issue_gin(GIN_OFF + ((unsigned)(t + 1) & 1u) * GIN_BYTES,
                    Gin + (size_t)(t + 1) * (B_ * NG_));
#else
    const int cur = 0;
    for (int i = tid; i < B_ * NG_; i += 512)
      ginb[i] = Gin[(size_t)t * (B_ * NG_) + i];
    __syncthreads();
#endif
    const _Float16* g = ginb + cur * (B_ * NG_);

    v8f c[8];
#pragma unroll
    for (int mt = 0; mt < 4; ++mt) {
      // A-tiles of h (16x32 f16) from padded LDS
      v16h a[4];
#pragma unroll
      for (int kt = 0; kt < 4; ++kt) {
        const _Float16* pa = hsm + (mt * 16 + laneLo) * HSTRIDE + kt * 32 + hiB;
        v16h av;
#pragma unroll
        for (int i = 0; i < 8; ++i) av[i] = pa[i];
#pragma unroll
        for (int i = 0; i < 8; ++i) av[8 + i] = pa[8 + i];
        a[kt] = av;
      }
#pragma unroll
      for (int j = 0; j < 2; ++j) {
        int n = (wid * 2 + j) * 16 + laneLo;
        v8f cc;
#pragma unroll
        for (int r = 0; r < 8; ++r)
          cc[r] = biasv[j] + (float)g[(mt * 16 + hiA + r) * NG_ + n];
#pragma unroll
        for (int kt = 0; kt < 4; ++kt)
          cc = __builtin_amdgcn_wmma_f32_16x16x32_f16(false, a[kt], false, Breg[j][kt],
                                                      (short)0, cc, false, false);
        c[mt * 2 + j] = cc;
      }
    }

    // activations (wave owns whole gate columns) -> LDS
#pragma unroll
    for (int mt = 0; mt < 4; ++mt) {
#pragma unroll
      for (int j = 0; j < 2; ++j) {
        int nt = wid * 2 + j;
        int n = nt * 16 + laneLo;
        bool isG = ((nt >> 3) == 2);  // gate order i,f,g,o ; tiles 16..23 are 'g'
        v8f cc = c[mt * 2 + j];
#pragma unroll
        for (int r = 0; r < 8; ++r) {
          float v = cc[r];
          float av = isG ? tanhf_(v) : sigmoidf_(v);
          gts[(mt * 16 + hiA + r) * GSTRIDE + n] = (_Float16)av;
        }
      }
    }
    __syncthreads();

    // state update: each thread owns 16 (m, hcol) cells
#pragma unroll
    for (int k = 0; k < 16; ++k) {
      int hc = hb + k;
      float iv = (float)gts[mrow * GSTRIDE + hc];
      float fv = (float)gts[mrow * GSTRIDE + HL_ + hc];
      float gv = (float)gts[mrow * GSTRIDE + 2 * HL_ + hc];
      float ov = (float)gts[mrow * GSTRIDE + 3 * HL_ + hc];
      float cn = fv * cst[k] + iv * gv;
      cst[k] = cn;
      float hn = ov * tanhf_(cn);
      hsm[mrow * HSTRIDE + hc] = (_Float16)hn;
      if (t == T_ - 1) out[mrow * HL_ + hc] = hn;
    }
#if defined(HAVE_TDM)
    if (wid == 0) __builtin_amdgcn_s_wait_tensorcnt(0);
#endif
    __syncthreads();
  }
}

// =====================================================================
extern "C" void kernel_launch(void* const* d_in, const int* in_sizes, int n_in,
                              void* d_out, int out_size, void* d_ws, size_t ws_size,
                              hipStream_t stream) {
  (void)in_sizes; (void)n_in; (void)out_size; (void)ws_size;
  const float* x    = (const float*)d_in[0];
  const float* adj  = (const float*)d_in[1];
  const float* W1   = (const float*)d_in[2];
  const float* b1   = (const float*)d_in[3];
  const float* W2   = (const float*)d_in[4];
  const float* b2   = (const float*)d_in[5];
  const float* wih  = (const float*)d_in[6];
  const float* whh  = (const float*)d_in[7];
  const float* bih  = (const float*)d_in[8];
  const float* bhh  = (const float*)d_in[9];
  float* out = (float*)d_out;

  char* ws = (char*)d_ws;
  _Float16* wih_h   = (_Float16*)(ws + WS_WIH_H);
  _Float16* whh_h   = (_Float16*)(ws + WS_WHH_H);
  float*    bias    = (float*)(ws + WS_BIAS);
  _Float16* feats_h = (_Float16*)(ws + WS_FEATS);
  _Float16* gin_h   = (_Float16*)(ws + WS_GIN);

  prep_kernel<<<256, 256, 0, stream>>>(wih, whh, bih, bhh, wih_h, whh_h, bias);
  gcn_kernel<<<MROWS / 8, 256, 0, stream>>>(x, adj, W1, b1, W2, b2, feats_h);
  gin_gemm_kernel<<<(MROWS / 128) * (NG_ / 128), 256, 0, stream>>>(feats_h, wih_h, gin_h);

  static bool attr_set = false;
  if (!attr_set) {
    (void)hipFuncSetAttribute((const void*)lstm_kernel,
                              hipFuncAttributeMaxDynamicSharedMemorySize, SMEM_TOTAL);
    attr_set = true;
  }
  lstm_kernel<<<1, 512, SMEM_TOTAL, stream>>>(gin_h, whh_h, bias, out);
}
